// GINNet_45028437131532
// MI455X (gfx1250) — matrix operand order repaired and checked by reading, not compiled, not run on hardware
//
#include <hip/hip_runtime.h>
#include <hip/hip_bf16.h>

typedef __attribute__((ext_vector_type(16))) _Float16 v16h;
typedef __attribute__((ext_vector_type(8)))  float    v8f;

// ---------------------------------------------------------------------------
// Kernel 1: build node features x = [pos | emb[z]] -> x[N,8]; also seed
// agg[N,8] = x so that after scatter-add agg == x + sum_{j->i} x_j directly.
// ---------------------------------------------------------------------------
__global__ void k_build(const float* __restrict__ pos, const int* __restrict__ z,
                        const float* __restrict__ emb,
                        float* __restrict__ x, float* __restrict__ agg, int n)
{
    int i = blockIdx.x * blockDim.x + threadIdx.x;
    if (i >= n) return;
    float v[8];
    v[0] = pos[(size_t)i * 3 + 0];
    v[1] = pos[(size_t)i * 3 + 1];
    v[2] = pos[(size_t)i * 3 + 2];
    const int zi = z[i];
#pragma unroll
    for (int j = 0; j < 5; ++j) v[3 + j] = emb[zi * 5 + j];
    float4 lo = make_float4(v[0], v[1], v[2], v[3]);
    float4 hi = make_float4(v[4], v[5], v[6], v[7]);
    float4* xp = (float4*)(x + (size_t)i * 8);
    float4* ap = (float4*)(agg + (size_t)i * 8);
    xp[0] = lo; xp[1] = hi;
    ap[0] = lo; ap[1] = hi;
}

// ---------------------------------------------------------------------------
// Kernel 2: zero pooled accumulator [G,64]
// ---------------------------------------------------------------------------
__global__ void k_zero(float* __restrict__ p, int n)
{
    int i = blockIdx.x * blockDim.x + threadIdx.x;
    if (i < n) p[i] = 0.0f;
}

// ---------------------------------------------------------------------------
// Kernel 3: edge scatter.  agg[dst] += x[src] (8 floats per edge).
// x fits in L2 (32 MB of 192 MB) -> gather + fp32 atomics are L2-resident.
// ---------------------------------------------------------------------------
__global__ void k_scatter(const int* __restrict__ src, const int* __restrict__ dst,
                          const float* __restrict__ x, float* __restrict__ agg, int ne)
{
    int e = blockIdx.x * blockDim.x + threadIdx.x;
    if (e >= ne) return;
    const int s = src[e];
    const int d = dst[e];
    const float4* xs = (const float4*)(x + (size_t)s * 8);
    const float4 a = xs[0];
    const float4 b = xs[1];
    float* o = agg + (size_t)d * 8;
    unsafeAtomicAdd(o + 0, a.x);
    unsafeAtomicAdd(o + 1, a.y);
    unsafeAtomicAdd(o + 2, a.z);
    unsafeAtomicAdd(o + 3, a.w);
    unsafeAtomicAdd(o + 4, b.x);
    unsafeAtomicAdd(o + 5, b.y);
    unsafeAtomicAdd(o + 6, b.z);
    unsafeAtomicAdd(o + 7, b.w);
}

// ---------------------------------------------------------------------------
// Kernel 4: fused MLP + pool.  One wave32 owns a 16-node tile.
//   layer1: [16,8(pad32)] @ [8(pad32),64]  -> 4 x v_wmma_f32_16x16x32_f16
//   layer2: [16,64]       @ [64,64]        -> 8 x v_wmma_f32_16x16x32_f16
// W1/W2 B-fragments are loop-invariant: built once per wave into registers.
// Layer1 output is re-laid-out (C-frag -> A-frag) through a per-wave LDS tile.
// ---------------------------------------------------------------------------
__global__ void __launch_bounds__(256) k_mlp(
    const float* __restrict__ hin,    // [N,8] = x + agg
    const float* __restrict__ W1,     // [8,64]
    const float* __restrict__ b1,     // [64]
    const float* __restrict__ W2,     // [64,64]
    const float* __restrict__ b2,     // [64]
    const int*   __restrict__ batch,  // [N]
    float*       __restrict__ pooled, // [G,64]
    int n_tiles)
{
    __shared__ __align__(16) _Float16 h1s[8][16 * 64];  // 8 waves * 2KB

    const int lane = threadIdx.x & 31;
    const int wid  = threadIdx.x >> 5;
    const int row  = lane & 15;
    const bool hi  = lane >= 16;
    const int mbase = hi ? 8 : 0;     // C/D fragment M base for this lane half
    const int koff  = hi ? 8 : 0;     // A fragment K sub-offset for this lane half

    // ---- B fragments (built once; reused across all tiles) ----
    // B layout (16x16x32): lanes 0-15 hold K=0..7,16..23 in elems 0..15 at N=lane;
    // lanes 16-31 hold K=8..15,24..31 at N=lane-16.
    v16h B1[4];
#pragma unroll
    for (int t = 0; t < 4; ++t) {
        v16h b = {};
        if (!hi) {
            const int col = t * 16 + lane;
#pragma unroll
            for (int j = 0; j < 8; ++j) b[j] = (_Float16)W1[j * 64 + col];
        }
        B1[t] = b;
    }
    v16h B2[4][2];
#pragma unroll
    for (int t = 0; t < 4; ++t) {
#pragma unroll
        for (int s = 0; s < 2; ++s) {
            v16h b;
            const int col = t * 16 + row;
            const int kb  = s * 32 + (hi ? 8 : 0);
#pragma unroll
            for (int j = 0; j < 16; ++j) b[j] = (_Float16)W2[(kb + j + (j >= 8 ? 8 : 0)) * 64 + col];
            B2[t][s] = b;
        }
    }

    const int gw = blockIdx.x * 8 + wid;
    const int nw = gridDim.x * 8;
    for (int tile = gw; tile < n_tiles; tile += nw) {
        const int base = tile * 16;

        // ---- A fragment, layer 1 (K=8 zero-padded to 32) ----
        // A layout: lane row = lane&15; elems 0..7 = K (lane<16 ? 0..7 : 8..15),
        // elems 8..15 = K 16.. -> all zero-pad here.
        v16h A1 = {};
        if (!hi) {
            const float4* hp = (const float4*)(hin + (size_t)(base + row) * 8);
            const float4 p0 = hp[0];
            const float4 p1 = hp[1];
            A1[0] = (_Float16)p0.x; A1[1] = (_Float16)p0.y;
            A1[2] = (_Float16)p0.z; A1[3] = (_Float16)p0.w;
            A1[4] = (_Float16)p1.x; A1[5] = (_Float16)p1.y;
            A1[6] = (_Float16)p1.z; A1[7] = (_Float16)p1.w;
        }

        // ---- layer 1: h1 = relu(A1 @ W1 + b1), staged to LDS ----
#pragma unroll
        for (int t = 0; t < 4; ++t) {
            v8f c = {};
            c = __builtin_amdgcn_wmma_f32_16x16x32_f16(
                    false, A1, false, B1[t], (short)0, c, false, false);
            const int col  = t * 16 + row;
            const float bs = b1[col];
#pragma unroll
            for (int v = 0; v < 8; ++v) {
                float val = c[v] + bs;
                val = val > 0.0f ? val : 0.0f;
                h1s[wid][(mbase + v) * 64 + col] = (_Float16)val;
            }
        }

        // ---- A fragments, layer 2 (from per-wave LDS tile) ----
        v16h A2[2];
#pragma unroll
        for (int s = 0; s < 2; ++s) {
            const _Float16* hrow = &h1s[wid][row * 64 + s * 32];
            v16h a;
#pragma unroll
            for (int j = 0; j < 8; ++j) a[j]     = hrow[koff + j];
#pragma unroll
            for (int j = 0; j < 8; ++j) a[8 + j] = hrow[16 + koff + j];
            A2[s] = a;
        }

        // graph ids for this lane's 8 output rows (same for all 4 n-tiles)
        int gidx[8];
#pragma unroll
        for (int v = 0; v < 8; ++v) gidx[v] = batch[base + mbase + v];

        // ---- layer 2 + relu + pooled atomic add ----
#pragma unroll
        for (int t = 0; t < 4; ++t) {
            v8f c = {};
            c = __builtin_amdgcn_wmma_f32_16x16x32_f16(
                    false, A2[0], false, B2[t][0], (short)0, c, false, false);
            c = __builtin_amdgcn_wmma_f32_16x16x32_f16(
                    false, A2[1], false, B2[t][1], (short)0, c, false, false);
            const int col  = t * 16 + row;
            const float bs = b2[col];
#pragma unroll
            for (int v = 0; v < 8; ++v) {
                float val = c[v] + bs;
                val = val > 0.0f ? val : 0.0f;
                unsafeAtomicAdd(&pooled[(size_t)gidx[v] * 64 + col], val);
            }
        }
    }
}

// ---------------------------------------------------------------------------
// Kernel 5: final FC  out[g] = pooled[g,:] . fcW + fcb
// ---------------------------------------------------------------------------
__global__ void k_fc(const float* __restrict__ pooled, const float* __restrict__ fcW,
                     const float* __restrict__ fcb, float* __restrict__ out, int g_count)
{
    int g = blockIdx.x * blockDim.x + threadIdx.x;
    if (g >= g_count) return;
    float acc = fcb[0];
    const float* p = pooled + (size_t)g * 64;
#pragma unroll
    for (int c = 0; c < 64; ++c) acc += p[c] * fcW[c];
    out[g] = acc;
}

// ---------------------------------------------------------------------------
extern "C" void kernel_launch(void* const* d_in, const int* in_sizes, int n_in,
                              void* d_out, int out_size, void* d_ws, size_t ws_size,
                              hipStream_t stream)
{
    const float* pos   = (const float*)d_in[0];
    const int*   z     = (const int*)  d_in[1];
    const int*   ei    = (const int*)  d_in[2];
    const int*   batch = (const int*)  d_in[3];
    const float* emb   = (const float*)d_in[4];
    const float* W1    = (const float*)d_in[5];
    const float* b1    = (const float*)d_in[6];
    const float* W2    = (const float*)d_in[7];
    const float* b2    = (const float*)d_in[8];
    const float* fcW   = (const float*)d_in[9];
    const float* fcb   = (const float*)d_in[10];
    float*       out   = (float*)d_out;

    const int N  = in_sizes[1];        // nodes (z count)
    const int NE = in_sizes[2] / 2;    // edges
    const int G  = out_size;           // graphs

    // workspace layout: x[N,8] | agg[N,8] | pooled[G,64]
    float* x      = (float*)d_ws;
    float* agg    = x + (size_t)N * 8;
    float* pooled = agg + (size_t)N * 8;

    k_build  <<<(N + 255) / 256, 256, 0, stream>>>(pos, z, emb, x, agg, N);
    k_zero   <<<(G * 64 + 255) / 256, 256, 0, stream>>>(pooled, G * 64);
    k_scatter<<<(NE + 255) / 256, 256, 0, stream>>>(ei, ei + NE, x, agg, NE);
    k_mlp    <<<1024, 256, 0, stream>>>(agg, W1, b1, W2, b2, batch, pooled, N / 16);
    k_fc     <<<(G + 255) / 256, 256, 0, stream>>>(pooled, fcW, fcb, out, G);
}
